// HMM_23295902613709
// MI455X (gfx1250) — matrix operand order repaired and checked by reading, not compiled, not run on hardware
//
#include <hip/hip_runtime.h>

// HMM forward on gfx1250: per-step log-matmul rewritten as bf16 WMMA GEMM.
//   Q[i,j] = exp(trans[i,j]-colmax_j)  (constant, bf16, staged in LDS)
//   P[b,i] = exp(alpha[b,i]-rowmax_b)  (bf16, L2-resident, rebuilt each step)
//   alpha' = emis + rowmax + colmax + log(P x Q)
// Persistent cooperative kernel, 16 blocks x 256 thr, grid barrier per phase.

#define B_   64
#define T_   1024
#define S_   512
#define NBLK 16
#define NTHR 256

typedef __attribute__((ext_vector_type(16))) __bf16 v16bf;
typedef __attribute__((ext_vector_type(8)))  float  v8f;

union ABFrag { uint4 u[2]; v16bf v; };

__device__ __forceinline__ unsigned short f2bf(float f) {
  unsigned u = __float_as_uint(f);
  u += 0x7FFFu + ((u >> 16) & 1u);     // round-to-nearest-even
  return (unsigned short)(u >> 16);
}

__device__ __forceinline__ void grid_barrier(unsigned* bar) {
  __syncthreads();
  if (threadIdx.x == 0) {
    volatile unsigned* gen = bar + 1;
    __threadfence();
    unsigned g = *gen;
    if (atomicAdd(bar, 1u) == (unsigned)(NBLK - 1)) {
      bar[0] = 0u;
      __threadfence();
      atomicAdd((unsigned*)(bar + 1), 1u);
    } else {
      while (*gen == g) __builtin_amdgcn_s_sleep(1);
    }
  }
  __syncthreads();
}

// One-time: column maxes of trans + Q = bf16(exp(trans - colmax)). 512 waves.
__global__ void hmm_prep(const float* __restrict__ trans,
                         unsigned short* __restrict__ Q,
                         float* __restrict__ mt,
                         unsigned* __restrict__ bar) {
  if (blockIdx.x == 0 && threadIdx.x == 0) { bar[0] = 0u; bar[1] = 0u; }
  int j    = blockIdx.x * 8 + (threadIdx.x >> 5);
  int lane = threadIdx.x & 31;
  int i0   = lane * 16;
  float v[16], m = -3.402823466e38f;
  for (int u = 0; u < 16; ++u) { v[u] = trans[(i0 + u) * S_ + j]; m = fmaxf(m, v[u]); }
  for (int d = 16; d >= 1; d >>= 1) m = fmaxf(m, __shfl_xor(m, d, 32));
  if (lane == 0) mt[j] = m;
  for (int u = 0; u < 16; ++u) Q[(i0 + u) * S_ + j] = f2bf(__expf(v[u] - m));
}

__global__ void __launch_bounds__(NTHR)
hmm_fwd(const int* __restrict__ obs, const float* __restrict__ emis,
        const float* __restrict__ prior,
        const unsigned short* __restrict__ Q, const float* __restrict__ mt,
        unsigned short* __restrict__ P, float* __restrict__ alpha,
        float* __restrict__ ma, unsigned* __restrict__ bar,
        float* __restrict__ out) {
  __shared__ unsigned short qlds[2 * S_ * 16];   // 32 KB: two 512x16 bf16 slabs

  const int tid  = threadIdx.x;
  const int lane = tid & 31;
  const int wid  = tid >> 5;
  const int gw   = blockIdx.x * (NTHR / 32) + wid;

  // Stage this block's two Q column-slabs (cols j0..j0+31) into LDS, once.
  const int j0blk = blockIdx.x * 32;
  for (int r = tid; r < 2 * S_; r += NTHR) {
    int slab = r >> 9, k = r & (S_ - 1);
    const uint4* src = (const uint4*)(Q + k * S_ + j0blk + slab * 16);
    uint4*       dst = (uint4*)(qlds + slab * (S_ * 16) + k * 16);
    dst[0] = src[0]; dst[1] = src[1];
  }

  // alpha0[b,s] = emis[obs[b,0], s] + prior[s]   (waves 0..63, 16 elems/lane)
  if (gw < B_) {
    int b = gw, o = obs[b * T_];
    const float4* er = (const float4*)(emis + o * S_ + lane * 16);
    const float4* pr = (const float4*)(prior + lane * 16);
    float4*       ar = (float4*)(alpha + b * S_ + lane * 16);
    for (int q = 0; q < 4; ++q) {
      float4 e = er[q], p = pr[q];
      ar[q] = make_float4(e.x + p.x, e.y + p.y, e.z + p.z, e.w + p.w);
    }
  }
  grid_barrier(bar);

  // Per-wave tile roles: block owns jTiles {2*blk, 2*blk+1}; 8 waves = 2x4 tiles.
  const int slab  = wid >> 2;          // which LDS Q slab (jTile within block)
  const int bTile = wid & 3;           // 4 b-tiles of 16 rows
  const int half  = lane >> 4;
  const int mlo   = lane & 15;
  const int jw    = j0blk + slab * 16 + mlo;   // output column (N = lane%16)
  const int arow  = bTile * 16 + mlo;          // A-fragment row in P
  const int kk    = mlo + (half << 4);         // B-fragment K row within chunk

  for (int t = 1; t < T_; ++t) {
    // ---- phase A: rowmax + P = bf16(exp(alpha - rowmax)) ----
    if (gw < B_) {
      int b = gw;
      const float4* ar = (const float4*)(alpha + b * S_ + lane * 16);
      float x[16], m = -3.402823466e38f;
      for (int q = 0; q < 4; ++q) {
        float4 f = ar[q];
        x[4*q+0] = f.x; x[4*q+1] = f.y; x[4*q+2] = f.z; x[4*q+3] = f.w;
      }
      for (int u = 0; u < 16; ++u) m = fmaxf(m, x[u]);
      for (int d = 16; d >= 1; d >>= 1) m = fmaxf(m, __shfl_xor(m, d, 32));
      unsigned pk[8];
      for (int u = 0; u < 8; ++u) {
        unsigned lo = f2bf(__expf(x[2*u]   - m));
        unsigned hi = f2bf(__expf(x[2*u+1] - m));
        pk[u] = lo | (hi << 16);
      }
      uint4* pp = (uint4*)(P + b * S_ + lane * 16);
      pp[0] = make_uint4(pk[0], pk[1], pk[2], pk[3]);
      pp[1] = make_uint4(pk[4], pk[5], pk[6], pk[7]);
      if (lane == 0) ma[b] = m;
    }
    grid_barrier(bar);

    // ---- phase B: R(16x16) = P(16x512) x Q(512x16), 16 chained WMMAs ----
    v8f acc = {};
    for (int c = 0; c < 16; ++c) {
      ABFrag a, bfrag;
      const uint4* ap = (const uint4*)(P + arow * S_ + c * 32 + half * 8);
      a.u[0] = ap[0];                  // K = 32c + 8h     .. +7
      a.u[1] = ap[2];                  // K = 32c + 16 + 8h .. +7
      const unsigned short* bq = qlds + slab * (S_ * 16) + (c * 32 + half * 8) * 16 + mlo;
      unsigned bw[8];
      for (int s = 0; s < 8; ++s) {
        int k0 = ((2 * s) & 7) + (((2 * s) >> 3) << 4);
        bw[s] = (unsigned)bq[k0 * 16] | ((unsigned)bq[(k0 + 1) * 16] << 16);
      }
      bfrag.u[0] = make_uint4(bw[0], bw[1], bw[2], bw[3]);
      bfrag.u[1] = make_uint4(bw[4], bw[5], bw[6], bw[7]);
      acc = __builtin_amdgcn_wmma_f32_16x16x32_bf16(
          false, a.v, false, bfrag.v, (short)0, acc, false, false);
    }
    // epilogue: alpha' = emis[obs_t] + ma + mt + log(R)
    float mtj = mt[jw];
    for (int r = 0; r < 8; ++r) {
      int b = bTile * 16 + (half << 3) + r;    // D: VGPR r -> M = r + 8*half
      int o = obs[b * T_ + t];
      alpha[b * S_ + jw] =
          emis[o * S_ + jw] + ma[b] + mtj + __logf(fmaxf(acc[r], 1e-37f));
    }
    grid_barrier(bar);
  }

  // ---- out[b] = logsumexp_s alpha[b,s] ----
  if (gw < B_) {
    int b = gw;
    const float4* ar = (const float4*)(alpha + b * S_ + lane * 16);
    float x[16], m = -3.402823466e38f;
    for (int q = 0; q < 4; ++q) {
      float4 f = ar[q];
      x[4*q+0] = f.x; x[4*q+1] = f.y; x[4*q+2] = f.z; x[4*q+3] = f.w;
    }
    for (int u = 0; u < 16; ++u) m = fmaxf(m, x[u]);
    for (int d = 16; d >= 1; d >>= 1) m = fmaxf(m, __shfl_xor(m, d, 32));
    float s = 0.f;
    for (int u = 0; u < 16; ++u) s += __expf(x[u] - m);
    for (int d = 16; d >= 1; d >>= 1) s += __shfl_xor(s, d, 32);
    if (lane == 0) out[b] = m + __logf(s);
  }
}

extern "C" void kernel_launch(void* const* d_in, const int* in_sizes, int n_in,
                              void* d_out, int out_size, void* d_ws, size_t ws_size,
                              hipStream_t stream) {
  const int*   obs   = (const int*)d_in[0];      // [64,1024]
  const float* emis  = (const float*)d_in[1];    // [1024,512]
  const float* trans = (const float*)d_in[2];    // [512,512]
  const float* prior = (const float*)d_in[3];    // [512]
  float* out = (float*)d_out;                    // [64]

  char* w = (char*)d_ws;                         // ~707 KB total
  unsigned short* Q     = (unsigned short*)(w);            // 512 KB bf16 exp(trans-colmax)
  unsigned short* P     = (unsigned short*)(w + 524288);   //  64 KB bf16 exp(alpha-rowmax)
  float*          alpha = (float*)(w + 589824);            // 128 KB
  float*          mt    = (float*)(w + 720896);            //   2 KB colmax
  float*          ma    = (float*)(w + 722944);            // 256 B rowmax
  unsigned*       bar   = (unsigned*)(w + 723200);         // grid barrier {count,gen}

  hmm_prep<<<64, 256, 0, stream>>>(trans, Q, mt, bar);
  hmm_fwd<<<NBLK, NTHR, 0, stream>>>(obs, emis, prior, Q, mt, P, alpha, ma, bar, out);
}
